// Attention_83502754169064
// MI455X (gfx1250) — matrix operand order repaired and checked by reading, not compiled, not run on hardware
//
#include <hip/hip_runtime.h>

// ---------------------------------------------------------------------------
// MI455X (gfx1250) attention pipeline, all matmuls via v_wmma_f32_16x16x32_bf16
//   B=4, T=2048, C=1024, H=16, CH=64
//   Workspace layout (bf16): Qp[BH,T,CH] | Kp[BH,T,CH] | Vt[BH,CH,T] | AO[B*T,C]
// Attention stages K/V tiles into LDS with GLOBAL_LOAD_ASYNC_TO_LDS_B128
// (ASYNCcnt) so the 4 waves of a block share one copy of each 8KB tile.
// ---------------------------------------------------------------------------

typedef __attribute__((ext_vector_type(16))) __bf16 v16bf;
typedef __attribute__((ext_vector_type(8)))  __bf16 v8bf;
typedef __attribute__((ext_vector_type(4)))  __bf16 v4bf;
typedef __attribute__((ext_vector_type(8)))  float  v8f;

#define WMMA_BF16(A, B, Cacc) \
  __builtin_amdgcn_wmma_f32_16x16x32_bf16(false, (A), false, (B), (short)0, (Cacc), false, false)

static __device__ __forceinline__ v16bf cat8(v8bf lo, v8bf hi) {
  union { v16bf v; v8bf h[2]; } u;
  u.h[0] = lo; u.h[1] = hi;
  return u.v;
}

static __device__ __forceinline__ v8f vzero() {
  v8f v;
#pragma unroll
  for (int i = 0; i < 8; ++i) v[i] = 0.0f;
  return v;
}

constexpr int kB  = 4;
constexpr int kT  = 2048;
constexpr int kC  = 1024;
constexpr int kH  = 16;
constexpr int kCH = 64;
constexpr int kM  = kB * kT;   // 8192 rows in the projection GEMMs

// ---------------------------------------------------------------------------
// GEMM: Out = A[M,K] @ W[N,K]^T + bias   (nn.Linear semantics, K = C = 1024)
//   OUT_MODE 0: bf16 out as [B*H, T, CH]         (Q / K projections)
//   OUT_MODE 1: bf16 out as [B*H, CH, T]         (V projection, transposed)
//   OUT_MODE 2: f32  out as [M, N] row-major     (final output projection)
//   A_F32: A operand is fp32 (converted to bf16 in LDS) vs already bf16.
// Block tile 128x128x32, 256 threads = 8 waves, wave tile 64x32 (4x2 WMMA).
// ---------------------------------------------------------------------------
template <int OUT_MODE, bool A_F32>
__global__ __launch_bounds__(256) void gemm_wmma_kernel(
    const void* __restrict__ Ap, const float* __restrict__ W,
    const float* __restrict__ bias, void* __restrict__ Out) {
  __shared__ __bf16 As[128][40];   // 32 k-cols + 8 pad, rows 16B-aligned
  __shared__ __bf16 Bs[128][40];

  const int tid  = threadIdx.x;
  const int lane = tid & 31;
  const int wid  = tid >> 5;
  const int lo   = lane & 15;
  const int hif  = lane >> 4;      // lane half (ISA fragment layouts key off this)
  const int kb   = hif * 8;        // A-frag k base: 0 (lanes 0-15) / 8 (lanes 16-31)
  const int wm   = wid >> 2;       // wave row  0..1  (64 rows each)
  const int wn   = wid & 3;        // wave col  0..3  (32 cols each)
  const int m0   = blockIdx.y * 128;
  const int n0   = blockIdx.x * 128;

  v8f acc[4][2];
#pragma unroll
  for (int mi = 0; mi < 4; ++mi)
#pragma unroll
    for (int nj = 0; nj < 2; ++nj) acc[mi][nj] = vzero();

  for (int k0 = 0; k0 < kC; k0 += 32) {
    __syncthreads();
    // ---- stage A tile (128x32) into LDS as bf16 ----
    if constexpr (A_F32) {
      const float* A = (const float*)Ap;
      const int r  = tid >> 3;
      const int c4 = (tid & 7) * 4;
#pragma unroll
      for (int i = 0; i < 4; ++i) {
        const int row = r + 32 * i;
        float4 f = *(const float4*)(A + (size_t)(m0 + row) * kC + k0 + c4);
        v4bf hv;
        hv[0] = (__bf16)f.x; hv[1] = (__bf16)f.y;
        hv[2] = (__bf16)f.z; hv[3] = (__bf16)f.w;
        *(v4bf*)&As[row][c4] = hv;
      }
    } else {
      const __bf16* A = (const __bf16*)Ap;
      const int r  = tid >> 2;
      const int c8 = (tid & 3) * 8;
#pragma unroll
      for (int i = 0; i < 2; ++i) {
        const int row = r + 64 * i;
        *(v8bf*)&As[row][c8] =
            *(const v8bf*)(A + (size_t)(m0 + row) * kC + k0 + c8);
      }
    }
    // ---- stage W tile (128 output-channels x 32 k) as bf16 ----
    {
      const int r  = tid >> 3;
      const int c4 = (tid & 7) * 4;
#pragma unroll
      for (int i = 0; i < 4; ++i) {
        const int row = r + 32 * i;
        float4 f = *(const float4*)(W + (size_t)(n0 + row) * kC + k0 + c4);
        v4bf hv;
        hv[0] = (__bf16)f.x; hv[1] = (__bf16)f.y;
        hv[2] = (__bf16)f.z; hv[3] = (__bf16)f.w;
        *(v4bf*)&Bs[row][c4] = hv;
      }
    }
    __syncthreads();

    // ---- fragments per ISA layouts, then 8 WMMAs ----
    v16bf af[4];
#pragma unroll
    for (int mi = 0; mi < 4; ++mi) {
      const int row = wm * 64 + mi * 16 + lo;           // A: lane = row
      af[mi] = cat8(*(const v8bf*)&As[row][kb],         // k = kb..kb+7
                    *(const v8bf*)&As[row][kb + 16]);   // k = kb+16..kb+23
    }
    v16bf wf[2];
#pragma unroll
    for (int nj = 0; nj < 2; ++nj) {
      const int n = wn * 32 + nj * 16 + lo;             // B: lane = col
      wf[nj] = cat8(*(const v8bf*)&Bs[n][kb],      // k = kb..kb+7
                    *(const v8bf*)&Bs[n][kb + 16]); // k = kb+16..kb+23
    }
#pragma unroll
    for (int mi = 0; mi < 4; ++mi)
#pragma unroll
      for (int nj = 0; nj < 2; ++nj)
        acc[mi][nj] = WMMA_BF16(af[mi], wf[nj], acc[mi][nj]);
  }

  // ---- epilogue: bias + layout-specific store ----
#pragma unroll
  for (int mi = 0; mi < 4; ++mi) {
#pragma unroll
    for (int nj = 0; nj < 2; ++nj) {
      const int n  = n0 + wn * 32 + nj * 16 + lo;
      const float bv = bias[n];
#pragma unroll
      for (int j = 0; j < 8; ++j) {
        const int m = m0 + wm * 64 + mi * 16 + j + 8 * hif;  // D: row j / j+8
        const float val = acc[mi][nj][j] + bv;
        if constexpr (OUT_MODE == 2) {
          ((float*)Out)[(size_t)m * kC + n] = val;
        } else {
          const int ob = m >> 11;          // m = b*T + t, T = 2048
          const int ot = m & (kT - 1);
          const int oh = n >> 6;           // n = h*CH + ch, CH = 64
          const int oc = n & (kCH - 1);
          size_t idx;
          if constexpr (OUT_MODE == 0)
            idx = ((size_t)(ob * kH + oh) * kT + ot) * kCH + oc;   // [BH,T,CH]
          else
            idx = ((size_t)(ob * kH + oh) * kCH + oc) * kT + ot;   // [BH,CH,T]
          ((__bf16*)Out)[idx] = (__bf16)val;
        }
      }
    }
  }
}

// ---------------------------------------------------------------------------
// Flash attention: grid (T/64, B*H); 128 threads = 4 waves, 16 query rows/wave.
// 64-key tiles: K tile [64 keys x 64 ch] and V tile [64 ch x 64 keys] staged
// into LDS once per block via GLOBAL_LOAD_ASYNC_TO_LDS_B128 (ASYNCcnt), then
// consumed as conflict-free ds_load_b128 fragments (row stride 144B).
// Per tile: 8 WMMA (Q.K^T) + one online-softmax pass + 8 WMMA (P.V).
// ---------------------------------------------------------------------------
__global__ __launch_bounds__(128) void attn_kernel(
    const __bf16* __restrict__ Qp, const __bf16* __restrict__ Kp,
    const __bf16* __restrict__ Vt, __bf16* __restrict__ Oout) {
  __shared__ __bf16 Ks[64][72];      // keys x channels (+8 pad)
  __shared__ __bf16 Vs[64][72];      // channels x keys (+8 pad)
  __shared__ __bf16 Pt[4][16][72];   // per-wave 16x64 P tile (+8 pad)

  const int tid  = threadIdx.x;
  const int lane = tid & 31;
  const int wid  = tid >> 5;
  const int lo   = lane & 15;
  const int hif  = lane >> 4;
  const int kb   = hif * 8;
  const int bh   = blockIdx.y;
  const int bb   = bh >> 4;          // batch
  const int hh   = bh & 15;          // head
  const int q0   = blockIdx.x * 64 + wid * 16;

  const __bf16* Qb = Qp + (size_t)bh * kT * kCH;
  const __bf16* Kb = Kp + (size_t)bh * kT * kCH;
  const __bf16* Vb = Vt + (size_t)bh * kCH * kT;

  // Q fragments (A-layout), kept in registers for the whole K sweep
  v16bf qf[2];
#pragma unroll
  for (int c = 0; c < 2; ++c) {
    const __bf16* p = Qb + (size_t)(q0 + lo) * kCH + c * 32 + kb;
    qf[c] = cat8(*(const v8bf*)p, *(const v8bf*)(p + 16));
  }

  v8f oacc[4];
#pragma unroll
  for (int nj = 0; nj < 4; ++nj) oacc[nj] = vzero();
  float mrow[8], lrow[8];
#pragma unroll
  for (int j = 0; j < 8; ++j) { mrow[j] = -3.0e38f; lrow[j] = 0.0f; }

  // LDS byte addresses (low 32 bits of a generic shared pointer = DS address)
  const unsigned ksBase = (unsigned)(size_t)&Ks[0][0];
  const unsigned vsBase = (unsigned)(size_t)&Vs[0][0];

  for (int kt = 0; kt < kT / 64; ++kt) {
    const int ktb = kt * 64;

    // ---- async stage: 512 x 16B chunks per tile, 4 per thread per tile ----
#pragma unroll
    for (int i = 0; i < 4; ++i) {
      const int chunk = tid + i * 128;       // 0..511
      const int row   = chunk >> 3;          // 0..63
      const int cc    = chunk & 7;           // 16B chunk within the 128B row
      const unsigned ldsK = ksBase + (unsigned)(row * 144 + cc * 16);
      const unsigned offK = (unsigned)(((ktb + row) * kCH + cc * 8) * 2);
      asm volatile("global_load_async_to_lds_b128 %0, %1, %2"
                   :: "v"(ldsK), "v"(offK), "s"(Kb) : "memory");
      const unsigned ldsV = vsBase + (unsigned)(row * 144 + cc * 16);
      const unsigned offV = (unsigned)((row * kT + ktb + cc * 8) * 2);
      asm volatile("global_load_async_to_lds_b128 %0, %1, %2"
                   :: "v"(ldsV), "v"(offV), "s"(Vb) : "memory");
    }
    asm volatile("s_wait_asynccnt 0" ::: "memory");
    __syncthreads();

    // ---- S = Q.K^T over 64 keys (4 n-tiles x 2 k-chunks = 8 WMMA) ----
    v8f s[4];
#pragma unroll
    for (int j2 = 0; j2 < 4; ++j2) s[j2] = vzero();
#pragma unroll
    for (int j2 = 0; j2 < 4; ++j2)
#pragma unroll
      for (int c = 0; c < 2; ++c) {
        const __bf16* p = &Ks[j2 * 16 + lo][c * 32 + kb];
        const v16bf kf = cat8(*(const v8bf*)p, *(const v8bf*)(p + 16));
        s[j2] = WMMA_BF16(qf[c], kf, s[j2]);
      }

    // ---- online softmax; half-wave reductions (xor 1,2,4,8 stay in-half) ----
    float alpha[8];
#pragma unroll
    for (int j = 0; j < 8; ++j) {
      const float x0 = s[0][j] * 0.125f;   // CH^-0.5 = 1/8
      const float x1 = s[1][j] * 0.125f;
      const float x2 = s[2][j] * 0.125f;
      const float x3 = s[3][j] * 0.125f;
      float mx = fmaxf(fmaxf(x0, x1), fmaxf(x2, x3));
      mx = fmaxf(mx, __shfl_xor(mx, 1, 32));
      mx = fmaxf(mx, __shfl_xor(mx, 2, 32));
      mx = fmaxf(mx, __shfl_xor(mx, 4, 32));
      mx = fmaxf(mx, __shfl_xor(mx, 8, 32));
      const float mn = fmaxf(mrow[j], mx);
      const float e0 = __expf(x0 - mn);
      const float e1 = __expf(x1 - mn);
      const float e2 = __expf(x2 - mn);
      const float e3 = __expf(x3 - mn);
      float rs = (e0 + e1) + (e2 + e3);
      rs += __shfl_xor(rs, 1, 32);
      rs += __shfl_xor(rs, 2, 32);
      rs += __shfl_xor(rs, 4, 32);
      rs += __shfl_xor(rs, 8, 32);
      const float al = __expf(mrow[j] - mn);
      lrow[j]  = lrow[j] * al + rs;
      mrow[j]  = mn;
      alpha[j] = al;
      const int r = j + 8 * hif;           // D-layout row -> LDS (A-layout src)
      Pt[wid][r][lo]      = (__bf16)e0;
      Pt[wid][r][16 + lo] = (__bf16)e1;
      Pt[wid][r][32 + lo] = (__bf16)e2;
      Pt[wid][r][48 + lo] = (__bf16)e3;
    }
    asm volatile("s_wait_dscnt 0" ::: "memory");  // wave-local P round-trip

    // P fragments in A-layout: row = lane, k-chunks of 32 keys
    v16bf pf[2];
#pragma unroll
    for (int c = 0; c < 2; ++c)
      pf[c] = cat8(*(const v8bf*)&Pt[wid][lo][c * 32 + kb],
                   *(const v8bf*)&Pt[wid][lo][c * 32 + kb + 16]);

    // ---- O += P.V  (4 ch-tiles x 2 key-chunks = 8 WMMA) ----
#pragma unroll
    for (int nj = 0; nj < 4; ++nj) {
#pragma unroll
      for (int j = 0; j < 8; ++j) oacc[nj][j] *= alpha[j];
#pragma unroll
      for (int c = 0; c < 2; ++c) {
        const __bf16* p = &Vs[nj * 16 + lo][c * 32 + kb];
        const v16bf vf = cat8(*(const v8bf*)p, *(const v8bf*)(p + 16));
        oacc[nj] = WMMA_BF16(pf[c], vf, oacc[nj]);
      }
    }
    __syncthreads();   // protect Ks/Vs before next tile's async overwrite
  }

  // normalize and store merged-head layout [B, T, H*CH] as bf16
#pragma unroll
  for (int j = 0; j < 8; ++j) {
    const float inv = 1.0f / lrow[j];
    const int t = q0 + j + 8 * hif;
    const size_t rowbase = ((size_t)bb * kT + t) * kC + hh * kCH;
#pragma unroll
    for (int nj = 0; nj < 4; ++nj)
      Oout[rowbase + nj * 16 + lo] = (__bf16)(oacc[nj][j] * inv);
  }
}

// ---------------------------------------------------------------------------
extern "C" void kernel_launch(void* const* d_in, const int* in_sizes, int n_in,
                              void* d_out, int out_size, void* d_ws,
                              size_t ws_size, hipStream_t stream) {
  (void)in_sizes; (void)n_in; (void)out_size; (void)ws_size;
  // setup_inputs order: k, q, v, Wk, bk, Wq, bq, Wv, bv, Wo, bo (all fp32)
  const float* kin = (const float*)d_in[0];
  const float* qin = (const float*)d_in[1];
  const float* vin = (const float*)d_in[2];
  const float* Wk  = (const float*)d_in[3];
  const float* bk  = (const float*)d_in[4];
  const float* Wq  = (const float*)d_in[5];
  const float* bq  = (const float*)d_in[6];
  const float* Wv  = (const float*)d_in[7];
  const float* bv  = (const float*)d_in[8];
  const float* Wo  = (const float*)d_in[9];
  const float* bo  = (const float*)d_in[10];

  char* ws = (char*)d_ws;
  const size_t SZ = (size_t)kB * kH * kT * kCH * sizeof(__bf16);  // 16 MiB
  __bf16* Qp = (__bf16*)(ws);
  __bf16* Kp = (__bf16*)(ws + SZ);
  __bf16* Vt = (__bf16*)(ws + 2 * SZ);
  __bf16* AO = (__bf16*)(ws + 3 * SZ);   // total 64 MiB of workspace

  const dim3 gg(kC / 128, kM / 128);     // (8, 64)
  const dim3 gb(256);

  gemm_wmma_kernel<0, true><<<gg, gb, 0, stream>>>(qin, Wq, bq, Qp);
  gemm_wmma_kernel<0, true><<<gg, gb, 0, stream>>>(kin, Wk, bk, Kp);
  gemm_wmma_kernel<1, true><<<gg, gb, 0, stream>>>(vin, Wv, bv, Vt);

  attn_kernel<<<dim3(kT / 64, kB * kH), dim3(128), 0, stream>>>(Qp, Kp, Vt, AO);

  gemm_wmma_kernel<2, false><<<gg, gb, 0, stream>>>(AO, Wo, bo, d_out);
}
